// GATBlock_592705487420
// MI455X (gfx1250) — matrix operand order repaired and checked by reading, not compiled, not run on hardware
//
#include <hip/hip_runtime.h>
#include <hip/hip_bf16.h>

// ---------------------------------------------------------------------------
// Problem constants (match the reference)
// ---------------------------------------------------------------------------
#define NN       50000
#define EE       1000000
#define IN_DIM   128
#define OUT_DIM  128
#define HH       4
#define CC       32           // OUT_DIM / H
#define EDGE_DIM 16
#define NEG_SLOPE 0.2f
#define EPS_SM   1e-16f
#define LN_EPS   1e-5f

#define M_TILES  (NN / 16)    // 3125

typedef __attribute__((ext_vector_type(16))) __bf16 v16bf;
typedef __attribute__((ext_vector_type(8)))  float  v8f;

// monotone mapping for float atomic-max on unsigned ints
__device__ __forceinline__ unsigned fmap(float f) {
    unsigned b = __float_as_uint(f);
    return (b & 0x80000000u) ? ~b : (b | 0x80000000u);
}
__device__ __forceinline__ float funmap(unsigned m) {
    unsigned b = (m & 0x80000000u) ? (m & 0x7FFFFFFFu) : ~m;
    return __uint_as_float(b);
}
#define FMAP_NEG_INF 0x007FFFFFu   // fmap(-inf)

__device__ __forceinline__ unsigned short f2bf(float f) {
    unsigned u = __float_as_uint(f);
    u += 0x7FFFu + ((u >> 16) & 1u);
    return (unsigned short)(u >> 16);
}

// ---------------------------------------------------------------------------
// 0a) fp32 -> bf16 (RNE) conversion (row-major, for the A operand / x)
// ---------------------------------------------------------------------------
__global__ void k_f32_to_bf16(const float* __restrict__ in,
                              unsigned short* __restrict__ out, int n) {
    int i = blockIdx.x * 256 + threadIdx.x;
    if (i >= n) return;
    out[i] = f2bf(in[i]);
}

// ---------------------------------------------------------------------------
// 0b) Pack a 128x128 f32 weight matrix directly into the WMMA B-operand
//     register layout (bf16):
//       packed[((nt*4 + kb)*32 + lane)*16 + j] = W[(kb*32 + (lane>>4)*8 + (j&7) + (j>>3)*16)*128
//                                                   + nt*16 + (lane&15)]
//     so the GEMM reads B with two b128 loads per k-step.
// ---------------------------------------------------------------------------
__global__ void k_pack_w(const float* __restrict__ W,
                         unsigned short* __restrict__ out) {
    int t = blockIdx.x * 256 + threadIdx.x;           // 0 .. 16383
    if (t >= 8 * 4 * 32 * 16) return;
    int j    = t & 15;
    int lane = (t >> 4) & 31;
    int kb   = (t >> 9) & 3;
    int nt   = t >> 11;
    int k    = kb * 32 + (lane >> 4) * 8 + (j & 7) + ((j >> 3) << 4);
    int col  = nt * 16 + (lane & 15);
    out[t] = f2bf(W[(size_t)k * OUT_DIM + col]);
}

// ---------------------------------------------------------------------------
// 1) Node GEMM: xl = x @ W_l, xr = x @ W_r  via v_wmma_f32_16x16x32_bf16.
//    grid = (ceil(M_TILES/8), 2), block = 256 (8 waves).
//    Each wave owns one 16-row M tile: A (16x128) lives in registers (loaded
//    once, 8x b128), then sweeps all 8 N tiles; B comes from the packed
//    layout as b128 loads. 32 WMMAs per wave.
// ---------------------------------------------------------------------------
__global__ __launch_bounds__(256)
void k_node_gemm(const unsigned short* __restrict__ xb,
                 const unsigned short* __restrict__ wpk_l,
                 const unsigned short* __restrict__ wpk_r,
                 float* __restrict__ xl, float* __restrict__ xr) {
    const unsigned short* wp  = blockIdx.y ? wpk_r : wpk_l;
    float*                dst = blockIdx.y ? xr : xl;

    const int wave = threadIdx.x >> 5;
    const int lane = threadIdx.x & 31;
    const int lo   = lane & 15;       // A row / B,C,D col within tile
    const int hi   = lane >> 4;       // half-wave select
    const int mt   = blockIdx.x * 8 + wave;
    if (mt >= M_TILES) return;
    const int mb = mt * 16;

    // ---- A: 16x128 bf16 slice in registers (4 k-blocks of 16x32).
    //      lane holds row mb+lo; vector elems 0..7 -> K=kb*32+hi*8+0..7,
    //      elems 8..15 -> K=kb*32+16+hi*8+0..7  (ISA 7.12.2, 16-bit A 16x32)
    v16bf A[4];
    #pragma unroll
    for (int kb = 0; kb < 4; ++kb) {
        union { uint4 q[2]; v16bf v; } au;
        const uint4* ap = reinterpret_cast<const uint4*>(
            xb + (size_t)(mb + lo) * IN_DIM + kb * 32 + hi * 8);
        au.q[0] = ap[0];   // elements 0..7
        au.q[1] = ap[2];   // elements 16..23 -> vector lanes 8..15
        A[kb] = au.v;
    }

    #pragma unroll
    for (int nt = 0; nt < 8; ++nt) {
        v8f acc = {};
        #pragma unroll
        for (int kb = 0; kb < 4; ++kb) {
            union { uint4 q[2]; v16bf v; } bu;
            const uint4* bp = reinterpret_cast<const uint4*>(
                wp + ((size_t)((nt * 4 + kb) * 32 + lane) << 4));
            bu.q[0] = bp[0];
            bu.q[1] = bp[1];
            acc = __builtin_amdgcn_wmma_f32_16x16x32_bf16(
                false, A[kb], false, bu.v, (short)0, acc, false, false);
        }
        // D 16x16 f32: lane col nt*16+lo, VGPR v -> row mb + hi*8 + v
        #pragma unroll
        for (int v = 0; v < 8; ++v)
            dst[(size_t)(mb + hi * 8 + v) * OUT_DIM + nt * 16 + lo] = acc[v];
    }
}

// ---------------------------------------------------------------------------
// 2) init: zero output accumulator, set max buffers to mapped(-inf), zero denom
// ---------------------------------------------------------------------------
__global__ void k_init(float* __restrict__ out, unsigned* __restrict__ mxm,
                       float* __restrict__ denom) {
    int i = blockIdx.x * 256 + threadIdx.x;
    if (i < NN * OUT_DIM) out[i] = 0.f;
    if (i < NN * HH) { mxm[i] = FMAP_NEG_INF; denom[i] = 0.f; }
}

// ---------------------------------------------------------------------------
// 3) Edge logits (fused ea = edge_attr @ W_e), leaky-relu, per-head dot with
//    att, plus atomic segment-max. One wave (32 lanes) per edge, 4 ch/lane.
// ---------------------------------------------------------------------------
__global__ __launch_bounds__(256)
void k_edge_logits(const int* __restrict__ src, const int* __restrict__ dst,
                   const float* __restrict__ edge_attr,
                   const float* __restrict__ We, const float* __restrict__ att,
                   const float* __restrict__ xl, const float* __restrict__ xr,
                   float* __restrict__ logits, unsigned* __restrict__ mxm) {
    __shared__ float sWe[EDGE_DIM * OUT_DIM];   // 8 KB
    __shared__ float sAtt[OUT_DIM];
    for (int i = threadIdx.x; i < EDGE_DIM * OUT_DIM; i += 256) sWe[i] = We[i];
    if (threadIdx.x < OUT_DIM) sAtt[threadIdx.x] = att[threadIdx.x];
    __syncthreads();

    const int wave = threadIdx.x >> 5, lane = threadIdx.x & 31;
    const int e = blockIdx.x * 8 + wave;
    if (e >= EE) return;
    const int s = src[e], d = dst[e];
    const int c0 = lane * 4;

    float ear[EDGE_DIM];
    #pragma unroll
    for (int k = 0; k < EDGE_DIM; ++k) ear[k] = edge_attr[(size_t)e * EDGE_DIM + k];

    const float4 vl = *reinterpret_cast<const float4*>(xl + (size_t)s * OUT_DIM + c0);
    const float4 vr = *reinterpret_cast<const float4*>(xr + (size_t)d * OUT_DIM + c0);
    const float ls[4] = {vl.x, vl.y, vl.z, vl.w};
    const float rs[4] = {vr.x, vr.y, vr.z, vr.w};

    float acc = 0.f;
    #pragma unroll
    for (int i = 0; i < 4; ++i) {
        float ea = 0.f;
        #pragma unroll
        for (int k = 0; k < EDGE_DIM; ++k) ea += ear[k] * sWe[k * OUT_DIM + c0 + i];
        float m = ls[i] + rs[i] + ea;
        m = (m > 0.f) ? m : m * NEG_SLOPE;              // leaky_relu
        acc += m * sAtt[c0 + i];
    }
    // reduce the 8 lanes belonging to one head (wave32)
    acc += __shfl_xor(acc, 1, 32);
    acc += __shfl_xor(acc, 2, 32);
    acc += __shfl_xor(acc, 4, 32);

    if ((lane & 7) == 0) {
        const int h = lane >> 3;
        logits[(size_t)e * HH + h] = acc;
        atomicMax(&mxm[(size_t)d * HH + h], fmap(acc));
    }
}

// ---------------------------------------------------------------------------
// 4) finalize segment max: unmap, -inf -> 0 (in place; uint buffer -> float)
// ---------------------------------------------------------------------------
__global__ void k_max_finalize(unsigned* __restrict__ mxm, float* __restrict__ mxf) {
    int i = blockIdx.x * 256 + threadIdx.x;
    if (i >= NN * HH) return;
    float f = funmap(mxm[i]);
    if (!isfinite(f)) f = 0.f;
    mxf[i] = f;
}

// ---------------------------------------------------------------------------
// 5) denom = segment_sum(exp(logit - mx[dst]))
// ---------------------------------------------------------------------------
__global__ void k_exp_denom(const int* __restrict__ dst,
                            const float* __restrict__ logits,
                            const float* __restrict__ mxf,
                            float* __restrict__ denom) {
    int t = blockIdx.x * 256 + threadIdx.x;
    if (t >= EE * HH) return;
    int e = t >> 2, h = t & 3;
    int d = dst[e];
    float ev = __expf(logits[t] - mxf[d * HH + h]);
    atomicAdd(&denom[d * HH + h], ev);
}

// ---------------------------------------------------------------------------
// 6) out[dst] += alpha * xl[src]   (wave per edge, 4 ch/lane, f32 L2 atomics)
// ---------------------------------------------------------------------------
__global__ __launch_bounds__(256)
void k_aggregate(const int* __restrict__ src, const int* __restrict__ dst,
                 const float* __restrict__ logits, const float* __restrict__ mxf,
                 const float* __restrict__ denom, const float* __restrict__ xl,
                 float* __restrict__ out) {
    const int wave = threadIdx.x >> 5, lane = threadIdx.x & 31;
    const int e = blockIdx.x * 8 + wave;
    if (e >= EE) return;
    const int s = src[e], d = dst[e];
    const int c0 = lane * 4;
    const int h  = lane >> 3;

    float lg = logits[(size_t)e * HH + h];
    float a  = __expf(lg - mxf[d * HH + h]) / (denom[d * HH + h] + EPS_SM);

    const float4 v = *reinterpret_cast<const float4*>(xl + (size_t)s * OUT_DIM + c0);
    float* op = out + (size_t)d * OUT_DIM + c0;
    atomicAdd(op + 0, a * v.x);
    atomicAdd(op + 1, a * v.y);
    atomicAdd(op + 2, a * v.z);
    atomicAdd(op + 3, a * v.w);
}

// ---------------------------------------------------------------------------
// 7) bias + LayerNorm + ReLU (wave per node, wave32 shfl reductions)
// ---------------------------------------------------------------------------
__global__ __launch_bounds__(256)
void k_layernorm(float* __restrict__ out, const float* __restrict__ bias,
                 const float* __restrict__ gamma, const float* __restrict__ beta) {
    const int wave = threadIdx.x >> 5, lane = threadIdx.x & 31;
    const int node = blockIdx.x * 8 + wave;
    if (node >= NN) return;
    const int c0 = lane * 4;
    float* row = out + (size_t)node * OUT_DIM;

    float4 v  = *reinterpret_cast<const float4*>(row + c0);
    float4 b4 = *reinterpret_cast<const float4*>(bias + c0);
    float x0 = v.x + b4.x, x1 = v.y + b4.y, x2 = v.z + b4.z, x3 = v.w + b4.w;

    float s  = x0 + x1 + x2 + x3;
    float ss = x0 * x0 + x1 * x1 + x2 * x2 + x3 * x3;
    #pragma unroll
    for (int m = 1; m < 32; m <<= 1) {
        s  += __shfl_xor(s,  m, 32);
        ss += __shfl_xor(ss, m, 32);
    }
    const float mean = s * (1.f / OUT_DIM);
    const float var  = ss * (1.f / OUT_DIM) - mean * mean;
    const float inv  = rsqrtf(var + LN_EPS);

    float4 g  = *reinterpret_cast<const float4*>(gamma + c0);
    float4 be = *reinterpret_cast<const float4*>(beta + c0);
    float4 r;
    r.x = fmaxf((x0 - mean) * inv * g.x + be.x, 0.f);
    r.y = fmaxf((x1 - mean) * inv * g.y + be.y, 0.f);
    r.z = fmaxf((x2 - mean) * inv * g.z + be.z, 0.f);
    r.w = fmaxf((x3 - mean) * inv * g.w + be.w, 0.f);
    *reinterpret_cast<float4*>(row + c0) = r;
}

// ---------------------------------------------------------------------------
// launcher
// ---------------------------------------------------------------------------
extern "C" void kernel_launch(void* const* d_in, const int* in_sizes, int n_in,
                              void* d_out, int out_size, void* d_ws, size_t ws_size,
                              hipStream_t stream) {
    const float* x         = (const float*)d_in[0];
    const int*   edge_idx  = (const int*)d_in[1];
    const float* edge_attr = (const float*)d_in[2];
    const float* W_l       = (const float*)d_in[3];
    const float* W_r       = (const float*)d_in[4];
    const float* W_e       = (const float*)d_in[5];
    const float* att       = (const float*)d_in[6];
    const float* bias      = (const float*)d_in[7];
    const float* ln_gamma  = (const float*)d_in[8];
    const float* ln_beta   = (const float*)d_in[9];
    float* out = (float*)d_out;

    const int* src = edge_idx;
    const int* dst = edge_idx + EE;

    // carve workspace (256B aligned)
    size_t off = 0;
    auto carve = [&](size_t bytes) {
        void* p = (char*)d_ws + off;
        off += (bytes + 255) & ~(size_t)255;
        return p;
    };
    unsigned short* xb     = (unsigned short*)carve((size_t)NN * IN_DIM * 2);
    unsigned short* wlb    = (unsigned short*)carve((size_t)IN_DIM * OUT_DIM * 2);
    unsigned short* wrb    = (unsigned short*)carve((size_t)IN_DIM * OUT_DIM * 2);
    float*          xl     = (float*)carve((size_t)NN * OUT_DIM * 4);
    float*          xr     = (float*)carve((size_t)NN * OUT_DIM * 4);
    float*          logits = (float*)carve((size_t)EE * HH * 4);
    unsigned*       mxm    = (unsigned*)carve((size_t)NN * HH * 4);
    float*          denom  = (float*)carve((size_t)NN * HH * 4);
    (void)ws_size;

    // 0) convert x to bf16; pack weights into WMMA B layout (bf16)
    k_f32_to_bf16<<<(NN * IN_DIM + 255) / 256, 256, 0, stream>>>(x, xb, NN * IN_DIM);
    k_pack_w<<<(IN_DIM * OUT_DIM + 255) / 256, 256, 0, stream>>>(W_l, wlb);
    k_pack_w<<<(IN_DIM * OUT_DIM + 255) / 256, 256, 0, stream>>>(W_r, wrb);

    // 1) node GEMMs on the WMMA pipe (wave per M-tile, A held in registers)
    k_node_gemm<<<dim3((M_TILES + 7) / 8, 2), 256, 0, stream>>>(xb, wlb, wrb, xl, xr);

    // 2) init accumulators
    k_init<<<(NN * OUT_DIM + 255) / 256, 256, 0, stream>>>(out, mxm, denom);

    // 3) fused edge logits + segment max
    k_edge_logits<<<EE / 8, 256, 0, stream>>>(src, dst, edge_attr, W_e, att,
                                              xl, xr, logits, mxm);

    // 4) finalize max (in place: uint -> float)
    k_max_finalize<<<(NN * HH + 255) / 256, 256, 0, stream>>>(mxm, (float*)mxm);

    // 5) softmax denominators
    k_exp_denom<<<(EE * HH + 255) / 256, 256, 0, stream>>>(dst, logits, (float*)mxm, denom);

    // 6) weighted scatter aggregation
    k_aggregate<<<EE / 8, 256, 0, stream>>>(src, dst, logits, (float*)mxm, denom, xl, out);

    // 7) bias + LayerNorm + ReLU
    k_layernorm<<<NN / 8, 256, 0, stream>>>(out, bias, ln_gamma, ln_beta);
}
